// MyModule_73589969649941
// MI455X (gfx1250) — matrix operand order repaired and checked by reading, not compiled, not run on hardware
//
#include <hip/hip_runtime.h>
#include <hip/hip_bf16.h>
#include <stdint.h>

// ---------------------------------------------------------------------------
// MI455X (gfx1250) split-bf16 WMMA MLP.
//   out = relu(relu(relu(X W1^T + b1) W2^T + b2) W3^T + b3)
//   thr = sum(out)/N_ROWS ; hi_mask = mask > thr
// All f32 operands pre-split into bf16 hi+lo planes (producer-side, once per
// element). GEMMs use 3x v_wmma_f32_16x16x32_bf16 (hi*hi + hi*lo + lo*hi) for
// ~fp32 accuracy at bf16 matrix rates; global->LDS uses async-to-LDS b128.
// ---------------------------------------------------------------------------

typedef __bf16 bf16;
typedef __attribute__((ext_vector_type(16))) __bf16 v16bf;
typedef __attribute__((ext_vector_type(4)))  __bf16 v4bf;
typedef __attribute__((ext_vector_type(8)))  float  v8f;
typedef __attribute__((ext_vector_type(4)))  int    v4i;

typedef __attribute__((address_space(1))) v4i* gv4i_p;  // global v4i*
typedef __attribute__((address_space(3))) v4i* lv4i_p;  // LDS v4i*

#if __has_builtin(__builtin_amdgcn_global_load_async_to_lds_b128)
#define ASYNC_LDS 1
#else
#define ASYNC_LDS 0
#endif

__device__ __forceinline__ void stage16(const void* g, void* l) {
#if ASYNC_LDS
  __builtin_amdgcn_global_load_async_to_lds_b128(
      (gv4i_p)(uintptr_t)g, (lv4i_p)(uint32_t)(uintptr_t)l, 0, 0);
#else
  *(float4*)l = *(const float4*)g;
#endif
}

__device__ __forceinline__ void wait_stage() {
#if ASYNC_LDS
#if __has_builtin(__builtin_amdgcn_s_wait_asynccnt)
  __builtin_amdgcn_s_wait_asynccnt(0);
#else
  asm volatile("s_wait_asynccnt 0" ::: "memory");
#endif
#endif
}

// Assemble a 16-element bf16 fragment from two 16B LDS chunks.
__device__ __forceinline__ v16bf load_frag(const bf16* p0, const bf16* p1) {
  union { v16bf v; uint4 q[2]; } u;
  u.q[0] = *(const uint4*)p0;
  u.q[1] = *(const uint4*)p1;
  return u.v;
}

// ---------------------------------------------------------------------------
// GEMM layer: relu(A[N,K] * W[H,K]^T + bias); A and W given as bf16 hi/lo
// planes. LAST=false: emit bf16 hi/lo planes (next layer's input).
// LAST=true : emit f32 C and accumulate total sum (for the threshold).
// Block tile 128(M) x 64(H), K-step 32; 8 waves as 4(M) x 2(H), each wave a
// 32x32 patch = 2x2 WMMA 16x16 tiles.
// ---------------------------------------------------------------------------
template <bool LAST>
__global__ __launch_bounds__(256) void mlp_layer(
    const bf16* __restrict__ AH,    // [Nrows, K] hi plane
    const bf16* __restrict__ AL,    // [Nrows, K] lo plane
    const bf16* __restrict__ WH,    // [H, K] hi plane
    const bf16* __restrict__ WL,    // [H, K] lo plane
    const float* __restrict__ bias, // [H]
    bf16*  __restrict__ CH,         // [Nrows, H] hi plane   (LAST=false)
    bf16*  __restrict__ CL,         // [Nrows, H] lo plane   (LAST=false)
    float* __restrict__ Cf,         // [Nrows, H] f32        (LAST=true)
    float* __restrict__ sumAcc,
    int K, int H)
{
  constexpr int MT = 128, NT = 64, KT = 32;
  __shared__ __align__(32) bf16 AsH[2][MT * KT];
  __shared__ __align__(32) bf16 AsL[2][MT * KT];
  __shared__ __align__(32) bf16 BsH[2][NT * KT];
  __shared__ __align__(32) bf16 BsL[2][NT * KT];
  __shared__ float red[256];

  const int tid  = threadIdx.x;
  const int lane = tid & 31;
  const int wave = tid >> 5;
  const int wm   = wave & 3;   // wave position along M (4)
  const int wn   = wave >> 2;  // wave position along H (2)
  const int l16  = lane & 15;
  const int half = lane >> 4;  // 0: lanes 0-15, 1: lanes 16-31
  const int m0   = blockIdx.x * MT;
  const int n0   = blockIdx.y * NT;

  auto stage = [&](int buf, int k0) {
    // A hi/lo: 128x32 bf16 = 512 x 16B chunks each -> 2 per thread per plane
#pragma unroll
    for (int i = 0; i < 2; ++i) {
      int idx = tid + i * 256;
      int row = idx >> 2, ch = idx & 3;              // 4 chunks per row
      size_t go = (size_t)(m0 + row) * K + k0 + ch * 8;
      stage16(AH + go, &AsH[buf][row * KT + ch * 8]);
      stage16(AL + go, &AsL[buf][row * KT + ch * 8]);
    }
    // W hi/lo: 64x32 bf16 = 256 x 16B chunks each -> 1 per thread per plane
    {
      int row = tid >> 2, ch = tid & 3;
      size_t go = (size_t)(n0 + row) * K + k0 + ch * 8;
      stage16(WH + go, &BsH[buf][row * KT + ch * 8]);
      stage16(WL + go, &BsL[buf][row * KT + ch * 8]);
    }
  };

  v8f acc[2][2];
#pragma unroll
  for (int mt = 0; mt < 2; ++mt)
#pragma unroll
    for (int nt = 0; nt < 2; ++nt)
#pragma unroll
      for (int r = 0; r < 8; ++r) acc[mt][nt][r] = 0.0f;

  stage(0, 0);
  wait_stage();
  __syncthreads();

  const int steps = K / KT;
  for (int s = 0; s < steps; ++s) {
    const int cur = s & 1;
    if (s + 1 < steps) stage(cur ^ 1, (s + 1) * KT);  // prefetch overlaps WMMA

    // A fragments (16-bit A 16x32 ISA layout): lane<16 -> K {0..7,16..23},
    // lane>=16 -> K {8..15,24..31}; two 16B chunks per fragment.
    v16bf aH[2], aL[2];
#pragma unroll
    for (int mt = 0; mt < 2; ++mt) {
      const int row = wm * 32 + mt * 16 + l16;
      const int a0  = row * KT + half * 8;
      aH[mt] = load_frag(&AsH[cur][a0], &AsH[cur][a0 + 16]);
      aL[mt] = load_frag(&AsL[cur][a0], &AsL[cur][a0 + 16]);
    }
    // B fragments (32x16 B layout): lane = N%16; lanes 0-15 K {0..7,16..23},
    // lanes 16-31 K {8..15,24..31}; two 16B chunks per fragment.
    v16bf bH[2], bL[2];
#pragma unroll
    for (int nt = 0; nt < 2; ++nt) {
      const int row = wn * 32 + nt * 16 + l16;
      const int b0  = row * KT + half * 8;
      bH[nt] = load_frag(&BsH[cur][b0], &BsH[cur][b0 + 16]);
      bL[nt] = load_frag(&BsL[cur][b0], &BsL[cur][b0 + 16]);
    }

    // split-precision accumulate: hi*hi + hi*lo + lo*hi
#pragma unroll
    for (int mt = 0; mt < 2; ++mt)
#pragma unroll
      for (int nt = 0; nt < 2; ++nt) {
        acc[mt][nt] = __builtin_amdgcn_wmma_f32_16x16x32_bf16(
            false, aH[mt], false, bH[nt], (short)0, acc[mt][nt], false, false);
        acc[mt][nt] = __builtin_amdgcn_wmma_f32_16x16x32_bf16(
            false, aH[mt], false, bL[nt], (short)0, acc[mt][nt], false, false);
        acc[mt][nt] = __builtin_amdgcn_wmma_f32_16x16x32_bf16(
            false, aL[mt], false, bH[nt], (short)0, acc[mt][nt], false, false);
      }

    wait_stage();
    __syncthreads();
  }

  // Epilogue. C/D layout: VGPR r -> lanes 0-15 M=r, lanes 16-31 M=r+8; N=l16.
  float ts = 0.0f;
#pragma unroll
  for (int mt = 0; mt < 2; ++mt) {
    const int rowB = m0 + wm * 32 + mt * 16 + (half ? 8 : 0);
#pragma unroll
    for (int nt = 0; nt < 2; ++nt) {
      const int col = n0 + wn * 32 + nt * 16 + l16;
      const float bv = bias[col];
#pragma unroll
      for (int r = 0; r < 8; ++r) {
        float v = acc[mt][nt][r] + bv;
        v = v > 0.0f ? v : 0.0f;
        const size_t idx = (size_t)(rowB + r) * H + col;
        if (LAST) {
          Cf[idx] = v;
          ts += v;
        } else {
          bf16 h = (bf16)v;        // split once at production
          CH[idx] = h;
          CL[idx] = (bf16)(v - (float)h);
        }
      }
    }
  }

  if (LAST) {
    red[tid] = ts;
    __syncthreads();
    for (int off = 128; off > 0; off >>= 1) {
      if (tid < off) red[tid] += red[tid + off];
      __syncthreads();
    }
    if (tid == 0) atomicAdd(sumAcc, red[0]);
  }
}

// f32 -> (bf16 hi, bf16 lo) split planes, 4 elements per thread
__global__ void split4_f32_bf16(const float4* __restrict__ w,
                                v4bf* __restrict__ hi, v4bf* __restrict__ lo,
                                int n4) {
  int i = blockIdx.x * 256 + threadIdx.x;
  if (i < n4) {
    float4 x = w[i];
    v4bf h, l;
    h[0] = (bf16)x.x; l[0] = (bf16)(x.x - (float)h[0]);
    h[1] = (bf16)x.y; l[1] = (bf16)(x.y - (float)h[1]);
    h[2] = (bf16)x.z; l[2] = (bf16)(x.z - (float)h[2]);
    h[3] = (bf16)x.w; l[3] = (bf16)(x.w - (float)h[3]);
    hi[i] = h;
    lo[i] = l;
  }
}

__global__ void zero_acc(float* acc) {
  if (threadIdx.x == 0 && blockIdx.x == 0) acc[0] = 0.0f;
}

__global__ void mask_threshold(const float4* __restrict__ mask,
                               const float* __restrict__ acc,
                               float4* __restrict__ out, int n4) {
  int i = blockIdx.x * 256 + threadIdx.x;
  if (i < n4) {
    const float thr = acc[0] * (1.0f / 262144.0f);  // mean of row sums
    float4 m = mask[i];
    out[i] = make_float4(m.x > thr ? 1.0f : 0.0f, m.y > thr ? 1.0f : 0.0f,
                         m.z > thr ? 1.0f : 0.0f, m.w > thr ? 1.0f : 0.0f);
  }
}

extern "C" void kernel_launch(void* const* d_in, const int* in_sizes, int n_in,
                              void* d_out, int out_size, void* d_ws,
                              size_t ws_size, hipStream_t stream) {
  (void)in_sizes; (void)n_in; (void)out_size; (void)ws_size;
  const float* X    = (const float*)d_in[0];
  const float* mask = (const float*)d_in[1];
  const float* W1   = (const float*)d_in[2];
  const float* b1   = (const float*)d_in[3];
  const float* W2   = (const float*)d_in[4];
  const float* b2   = (const float*)d_in[5];
  const float* W3   = (const float*)d_in[6];
  const float* b3   = (const float*)d_in[7];
  float* out = (float*)d_out;

  constexpr int NR = 512 * 512, DIN = 512, H1 = 1024, H2 = 1024, DOUT = 512;

  // ws layout: [acc | weight planes (<16MB) | X planes | h1 planes | h2 planes]
  char* ws   = (char*)d_ws;
  float* acc = (float*)ws;
  bf16* W1H  = (bf16*)(ws + 256);
  bf16* W1L  = W1H + (size_t)H1 * DIN;
  bf16* W2H  = W1L + (size_t)H1 * DIN;
  bf16* W2L  = W2H + (size_t)H2 * H1;
  bf16* W3H  = W2L + (size_t)H2 * H1;
  bf16* W3L  = W3H + (size_t)DOUT * H2;
  bf16* XH   = (bf16*)(ws + (16ull << 20));
  bf16* XL   = XH + (size_t)NR * DIN;
  bf16* h1H  = XL + (size_t)NR * DIN;
  bf16* h1L  = h1H + (size_t)NR * H1;
  bf16* h2H  = h1L + (size_t)NR * H1;
  bf16* h2L  = h2H + (size_t)NR * H2;

  zero_acc<<<1, 64, 0, stream>>>(acc);
  split4_f32_bf16<<<(H1 * DIN / 4) / 256, 256, 0, stream>>>(
      (const float4*)W1, (v4bf*)W1H, (v4bf*)W1L, H1 * DIN / 4);
  split4_f32_bf16<<<(H2 * H1 / 4) / 256, 256, 0, stream>>>(
      (const float4*)W2, (v4bf*)W2H, (v4bf*)W2L, H2 * H1 / 4);
  split4_f32_bf16<<<(DOUT * H2 / 4) / 256, 256, 0, stream>>>(
      (const float4*)W3, (v4bf*)W3H, (v4bf*)W3L, DOUT * H2 / 4);
  split4_f32_bf16<<<(int)(((size_t)NR * DIN / 4) / 256), 256, 0, stream>>>(
      (const float4*)X, (v4bf*)XH, (v4bf*)XL, (int)((size_t)NR * DIN / 4));

  dim3 blk(256);
  mlp_layer<false><<<dim3(NR / 128, H1 / 64), blk, 0, stream>>>(
      XH, XL, W1H, W1L, b1, h1H, h1L, nullptr, nullptr, DIN, H1);
  mlp_layer<false><<<dim3(NR / 128, H2 / 64), blk, 0, stream>>>(
      h1H, h1L, W2H, W2L, b2, h2H, h2L, nullptr, nullptr, H1, H2);
  mlp_layer<true><<<dim3(NR / 128, DOUT / 64), blk, 0, stream>>>(
      h2H, h2L, W3H, W3L, b3, nullptr, nullptr, out, acc, H2, DOUT);

  const size_t NM = 512ull * 512ull * 512ull;
  mask_threshold<<<(int)(NM / 4 / 256), 256, 0, stream>>>(
      (const float4*)mask, acc, (float4*)(out + (size_t)NR * DOUT),
      (int)(NM / 4));
}
